// LSTM_87351044866551
// MI455X (gfx1250) — matrix operand order repaired and checked by reading, not compiled, not run on hardware
//
#include <hip/hip_runtime.h>
#include <hip/hip_bf16.h>

typedef __bf16 bf16_t;
typedef __attribute__((ext_vector_type(16))) __bf16 v16bf;
typedef __attribute__((ext_vector_type(8)))  __bf16 v8bf;
typedef __attribute__((ext_vector_type(8)))  float  v8f;
typedef __attribute__((ext_vector_type(4)))  float  v4f;

#define WMMA_BF16(A, Bm, C) \
    __builtin_amdgcn_wmma_f32_16x16x32_bf16(false, (A), false, (Bm), (short)0, (C), false, false)

// Problem constants
#define BB 256
#define TT 128
#define II 2048
#define HH 128
#define G4 512   // 4*H

// ---------------------------------------------------------------------------
// Prep: transpose + f32->bf16 convert weight matrices
// ---------------------------------------------------------------------------
__global__ void prep_wiT(const float* __restrict__ Wi, bf16_t* __restrict__ WiT) {
    int n = blockIdx.x * 256 + threadIdx.x;      // 0 .. 512*2048-1
    int g = n >> 11, i = n & 2047;
    WiT[n] = (bf16_t)Wi[i * G4 + g];             // WiT[g][i]
}

__global__ void prep_uhT(const float* __restrict__ Uh, bf16_t* __restrict__ UhT) {
    int n = blockIdx.x * 256 + threadIdx.x;      // 0 .. 512*128-1
    int g = n >> 7, k = n & 127;
    UhT[n] = (bf16_t)Uh[k * G4 + g];             // UhT[g][k]
}

// ---------------------------------------------------------------------------
// GEMM1: xg[B*T, 4H] = x[B*T, I] @ Wi[I, 4H] + b
// Each wave: 16 rows x 128 cols. Software-pipelined:
//   - A (x) f32 loads for k+32 issued one iteration early
//   - all 8 B tiles loaded, sched_barrier, then WMMA chain drains with
//     staggered s_wait_loadcnt so loads overlap the whole chain.
// ---------------------------------------------------------------------------
__global__ __launch_bounds__(256) void gemm_xg(
    const float* __restrict__ x, const bf16_t* __restrict__ WiT,
    const float* __restrict__ bias, float* __restrict__ xg)
{
    const int lane  = threadIdx.x & 31;
    const int wave  = threadIdx.x >> 5;            // 0..7
    const int m0    = blockIdx.x * 128 + wave * 16;
    const int nbase = blockIdx.y * 128;
    const int lmod  = lane & 15;
    const int lhi   = lane >> 4;                   // 0 or 1
    const int khalf = lhi * 16;

    const float* xrow = x + (size_t)(m0 + lmod) * II;

    // per-lane base pointers for the 8 B tiles (K-contiguous rows of WiT)
    const bf16_t* brow[8];
#pragma unroll
    for (int j = 0; j < 8; ++j)
        brow[j] = WiT + (size_t)(nbase + j * 16 + lmod) * II + lhi * 16;

    v8f acc[8];
#pragma unroll
    for (int j = 0; j < 8; ++j) {
        float bv = bias[nbase + j * 16 + lmod];
#pragma unroll
        for (int r = 0; r < 8; ++r) acc[j][r] = bv;
    }

    // ---- A pipeline prologue: load k0 = 0 -----------------------------
    v4f p0 = *(const v4f*)(xrow + khalf);
    v4f p1 = *(const v4f*)(xrow + khalf + 4);
    v4f p2 = *(const v4f*)(xrow + 8 + khalf);
    v4f p3 = *(const v4f*)(xrow + 8 + khalf + 4);

    for (int k0 = 0; k0 < II; k0 += 32) {
        // prefetch the x stream ahead (global_prefetch_b8)
        __builtin_prefetch((const void*)(xrow + k0 + 256), 0, 3);

        // convert the A tile loaded last iteration
        v16bf a;
#pragma unroll
        for (int e = 0; e < 4; ++e) {
            a[e]      = (bf16_t)p0[e];
            a[4 + e]  = (bf16_t)p1[e];
            a[8 + e]  = (bf16_t)p2[e];
            a[12 + e] = (bf16_t)p3[e];
        }

        // issue all 8 B-tile loads (16x global_load_b128)
        v16bf bm[8];
#pragma unroll
        for (int j = 0; j < 8; ++j)
            bm[j] = *(const v16bf*)(brow[j] + k0);

        // issue next iteration's A loads so they hide behind the WMMAs
        if (k0 + 32 < II) {
            const float* xn = xrow + k0 + 32;
            p0 = *(const v4f*)(xn + khalf);
            p1 = *(const v4f*)(xn + khalf + 4);
            p2 = *(const v4f*)(xn + 8 + khalf);
            p3 = *(const v4f*)(xn + 8 + khalf + 4);
        }

        // nothing crosses: forces 8 live B buffers + staggered load waits
        __builtin_amdgcn_sched_barrier(0);

#pragma unroll
        for (int j = 0; j < 8; ++j)
            acc[j] = WMMA_BF16(a, bm[j], acc[j]);
    }

#pragma unroll
    for (int j = 0; j < 8; ++j) {
        int col = nbase + j * 16 + lmod;
#pragma unroll
        for (int r = 0; r < 8; ++r) {
            int row = m0 + r + 8 * lhi;
            xg[(size_t)row * G4 + col] = acc[j][r];
        }
    }
}

// ---------------------------------------------------------------------------
// Recurrence: 16 workgroups, each owns 16 batch rows; full T loop in-kernel
// with only workgroup barriers. h (bf16), c (f32), z (f32) AND the full
// UhT weight matrix live in LDS (172 KB of the 320 KB/WGP budget).
// ---------------------------------------------------------------------------
__global__ __launch_bounds__(256) void lstm_scan(
    const float* __restrict__ xg, const bf16_t* __restrict__ UhT,
    float* __restrict__ hfin)
{
    __shared__ __align__(16) bf16_t sh_h[16 * HH];   //   4 KB
    __shared__ float  sh_c[16 * HH];                 //   8 KB
    __shared__ float  sh_z[16 * G4];                 //  32 KB
    __shared__ __align__(16) bf16_t sh_u[G4 * HH];   // 128 KB

    const int tid   = threadIdx.x;
    const int lane  = tid & 31;
    const int wave  = tid >> 5;                      // 0..7
    const int lmod  = lane & 15;
    const int lhi   = lane >> 4;
    const int khalf = lhi * 16;
    const int b0    = blockIdx.x * 16;

    // stage UhT into LDS once (16-byte chunks, coalesced)
    {
        const uint4* src = (const uint4*)UhT;
        uint4* dst = (uint4*)sh_u;
        for (int i = tid; i < (G4 * HH * 2) / 16; i += 256) dst[i] = src[i];
    }
    for (int i = tid; i < 16 * HH; i += 256) { sh_h[i] = (bf16_t)0.f; sh_c[i] = 0.f; }
    __syncthreads();

    for (int t = 0; t < TT; ++t) {
        // ---- phase 1: z = xg[:, t, :] + h @ Uh -------------------------
        v16bf a[4];
#pragma unroll
        for (int kk = 0; kk < 4; ++kk) {
            const bf16_t* hrow = &sh_h[lmod * HH + kk * 32 + khalf];
            v8bf lo = *(const v8bf*)hrow;
            v8bf hi = *(const v8bf*)(hrow + 8);
#pragma unroll
            for (int e = 0; e < 8; ++e) { a[kk][e] = lo[e]; a[kk][8 + e] = hi[e]; }
        }
#pragma unroll
        for (int jt = 0; jt < 4; ++jt) {
            int n0 = wave * 64 + jt * 16;
            // preload the 4 K-chunk B tiles for this output tile
            v16bf bm[4];
#pragma unroll
            for (int kk = 0; kk < 4; ++kk)
                bm[kk] = *(const v16bf*)&sh_u[(n0 + lmod) * HH + kk * 32 + lhi * 16];
            v8f acc;
#pragma unroll
            for (int r = 0; r < 8; ++r) {
                int row = b0 + r + 8 * lhi;
                acc[r] = xg[((size_t)row * TT + t) * G4 + n0 + lmod];
            }
#pragma unroll
            for (int kk = 0; kk < 4; ++kk)
                acc = WMMA_BF16(a[kk], bm[kk], acc);
#pragma unroll
            for (int r = 0; r < 8; ++r)
                sh_z[(r + 8 * lhi) * G4 + n0 + lmod] = acc[r];
        }
        __syncthreads();

        // ---- phase 2: gate update (i, f, g, o) -------------------------
        {
            int row = tid >> 4;            // 0..15
            int jb  = (tid & 15) * 8;      // 0..120
#pragma unroll
            for (int u = 0; u < 8; ++u) {
                int j = jb + u;
                float zi = sh_z[row * G4 + j];
                float zf = sh_z[row * G4 + 128 + j];
                float zg = sh_z[row * G4 + 256 + j];
                float zo = sh_z[row * G4 + 384 + j];
                float c  = sh_c[row * HH + j];
                float ig = 1.f / (1.f + __expf(-zi));
                float fg = 1.f / (1.f + __expf(-zf));
                float og = 1.f / (1.f + __expf(-zo));
                float gg = tanhf(zg);
                float cn = fg * c + ig * gg;
                float hn = og * tanhf(cn);
                sh_c[row * HH + j] = cn;
                sh_h[row * HH + j] = (bf16_t)hn;
                if (t == TT - 1) hfin[(size_t)(b0 + row) * HH + j] = hn;
            }
        }
        __syncthreads();
    }
}

// ---------------------------------------------------------------------------
// Output head: out[256,2] = h_f @ Wout + bout  (tiny, scalar)
// ---------------------------------------------------------------------------
__global__ void head_kernel(const float* __restrict__ hfin,
                            const float* __restrict__ Wout,
                            const float* __restrict__ bout,
                            float* __restrict__ out)
{
    int gid = blockIdx.x * blockDim.x + threadIdx.x;
    if (gid >= BB * 2) return;
    int b = gid >> 1, o = gid & 1;
    float s = bout[o];
#pragma unroll 4
    for (int j = 0; j < HH; ++j) s += hfin[b * HH + j] * Wout[j * 2 + o];
    out[gid] = s;
}

// ---------------------------------------------------------------------------
extern "C" void kernel_launch(void* const* d_in, const int* in_sizes, int n_in,
                              void* d_out, int out_size, void* d_ws, size_t ws_size,
                              hipStream_t stream)
{
    (void)in_sizes; (void)n_in; (void)out_size; (void)ws_size;
    const float* x    = (const float*)d_in[0];
    const float* Wi   = (const float*)d_in[1];
    const float* Uh   = (const float*)d_in[2];
    const float* b    = (const float*)d_in[3];
    const float* Wout = (const float*)d_in[4];
    const float* bout = (const float*)d_in[5];
    float* out = (float*)d_out;

    char* ws = (char*)d_ws;
    const size_t XG_BYTES  = (size_t)BB * TT * G4 * sizeof(float);   // 64 MB
    const size_t WIT_BYTES = (size_t)G4 * II * sizeof(bf16_t);       // 2 MB
    const size_t UHT_BYTES = (size_t)G4 * HH * sizeof(bf16_t);       // 128 KB

    float*  xg   = (float*)(ws);
    bf16_t* WiT  = (bf16_t*)(ws + XG_BYTES);
    bf16_t* UhT  = (bf16_t*)(ws + XG_BYTES + WIT_BYTES);
    float*  hfin = (float*)(ws + XG_BYTES + WIT_BYTES + UHT_BYTES);

    prep_wiT<<<(G4 * II) / 256, 256, 0, stream>>>(Wi, WiT);
    prep_uhT<<<(G4 * HH) / 256, 256, 0, stream>>>(Uh, UhT);
    gemm_xg<<<dim3((BB * TT) / 128, G4 / 128), 256, 0, stream>>>(x, WiT, b, xg);
    lstm_scan<<<BB / 16, 256, 0, stream>>>(xg, UhT, hfin);
    head_kernel<<<2, 256, 0, stream>>>(hfin, Wout, bout, out);
}
